// StyleSwinBlock_67637144977829
// MI455X (gfx1250) — matrix operand order repaired and checked by reading, not compiled, not run on hardware
//
#include <hip/hip_runtime.h>
#include <cstddef>
#include <cstdint>

typedef __bf16 bf16;
typedef __attribute__((ext_vector_type(16))) __bf16 v16bf;
typedef __attribute__((ext_vector_type(8)))  __bf16 v8bf;
typedef __attribute__((ext_vector_type(8)))  float  v8f;

#define RES_ 64
#define DIM_ 512
#define NTOK 4096   // RES*RES
#define BATCH 8
#define MTOT 32768  // BATCH*NTOK

__device__ __forceinline__ v16bf make_v16(v8bf lo, v8bf hi) {
  v16bf r;
#pragma unroll
  for (int i = 0; i < 8; ++i) { r[i] = lo[i]; r[8 + i] = hi[i]; }
  return r;
}
__device__ __forceinline__ v8f zero8() {
  v8f r;
#pragma unroll
  for (int i = 0; i < 8; ++i) r[i] = 0.0f;
  return r;
}

// CDNA5 direct global->LDS async copy (16B per lane), tracked by ASYNCcnt.
__device__ __forceinline__ void async_g2l_b128(uint32_t lds_off, const void* g) {
  asm volatile("global_load_async_to_lds_b128 %0, %1, off"
               :: "v"(lds_off), "v"(g) : "memory");
}
__device__ __forceinline__ void wait_async_le4() {
  asm volatile("s_wait_asynccnt 0x4" ::: "memory");
}
__device__ __forceinline__ void wait_async_0() {
  asm volatile("s_wait_asynccnt 0x0" ::: "memory");
}

// ---------------- style modulation: s[b][o] = sum_k style[b][k]*w[o][k]*scale + bias[o]
__global__ __launch_bounds__(256) void style_kernel(
    const float* __restrict__ style, const float* __restrict__ w,
    const float* __restrict__ bias, float* __restrict__ sout) {
  const int b = blockIdx.x;
  const float scale = 0.04419417382415922f;  // 1/sqrt(512)
  const float* st = style + b * 512;
#pragma unroll
  for (int oo = 0; oo < 4; ++oo) {
    int o = threadIdx.x + oo * 256;
    const float* wr = w + (size_t)o * 512;
    float acc = 0.f;
    for (int k = 0; k < 512; ++k) acc += st[k] * wr[k];
    sout[b * 1024 + o] = acc * scale + bias[o];
  }
}

// ---------------- per-(b,c) mean / rstd over N=4096 tokens
__global__ __launch_bounds__(256) void stats_kernel(
    const float* __restrict__ X, float* __restrict__ meanb, float* __restrict__ rstdb) {
  const int c = (blockIdx.x & 1) * 256 + threadIdx.x;
  const int b = blockIdx.x >> 1;
  const float* p = X + (size_t)b * NTOK * DIM_ + c;
  float s = 0.f, sq = 0.f;
  for (int n = 0; n < NTOK; ++n) {
    float v = p[(size_t)n * DIM_];
    s += v; sq += v * v;
  }
  float mean = s * (1.0f / NTOK);
  float var = sq * (1.0f / NTOK) - mean * mean;
  meanb[b * DIM_ + c] = mean;
  rstdb[b * DIM_ + c] = rsqrtf(var + 1e-5f);
}

// ---------------- adain normalize -> bf16
__global__ __launch_bounds__(256) void norm_kernel(
    const float* __restrict__ X, const float* __restrict__ meanb,
    const float* __restrict__ rstdb, const float* __restrict__ s,
    bf16* __restrict__ out) {
  size_t i = (size_t)blockIdx.x * 256 + threadIdx.x;
  int c = (int)(i & 511);
  int b = (int)(i >> 21);  // i / (4096*512)
  float g = s[b * 1024 + c], be = s[b * 1024 + 512 + c];
  float m = meanb[b * DIM_ + c], r = rstdb[b * DIM_ + c];
  out[i] = (bf16)(((X[i] - m) * r) * g + be);
}

// ---------------- f32 -> bf16
__global__ __launch_bounds__(256) void cvt_kernel(const float* __restrict__ in,
                                                  bf16* __restrict__ out, int n) {
  int i = blockIdx.x * 256 + threadIdx.x;
  if (i < n) out[i] = (bf16)in[i];
}

// ---------------- generic bf16 WMMA GEMM: out[M,N] = A[M,K] * W[N,K]^T + bias (+gelu / +residual)
// Block tile 128x128, 8 waves each 64x32, K-step 32.
// Double-buffered async global->LDS staging (ASYNCcnt), WMMA consume.
template <int KT, bool GELU, int OUTF>  // OUTF: 0 = bf16 out, 1 = f32 out (+residual)
__global__ __launch_bounds__(256) void gemm_kernel(
    const bf16* __restrict__ A, const bf16* __restrict__ W,
    const float* __restrict__ bias, const float* __restrict__ res,
    void* __restrict__ outp, int Nout) {
  __shared__ __align__(32) bf16 As[2][128 * 32];
  __shared__ __align__(32) bf16 Bs[2][128 * 32];
  const int tid = threadIdx.x;
  const int lane = tid & 31, wid = tid >> 5;
  const int wm = wid >> 2, wn = wid & 3;
  const int khalf = lane >> 4, l16 = lane & 15;
  const int m0 = blockIdx.y * 128;
  const int n0 = blockIdx.x * 128;

  // this thread's two (row, seg) staging slots
  const int row0 = tid >> 2, seg0 = tid & 3;
  const int row1 = (tid + 256) >> 2, seg1 = (tid + 256) & 3;

  auto issue_tile = [&](int k0, int buf) {
    async_g2l_b128((uint32_t)(uintptr_t)(&As[buf][row0 * 32 + seg0 * 8]),
                   A + (size_t)(m0 + row0) * KT + k0 + seg0 * 8);
    async_g2l_b128((uint32_t)(uintptr_t)(&Bs[buf][row0 * 32 + seg0 * 8]),
                   W + (size_t)(n0 + row0) * KT + k0 + seg0 * 8);
    async_g2l_b128((uint32_t)(uintptr_t)(&As[buf][row1 * 32 + seg1 * 8]),
                   A + (size_t)(m0 + row1) * KT + k0 + seg1 * 8);
    async_g2l_b128((uint32_t)(uintptr_t)(&Bs[buf][row1 * 32 + seg1 * 8]),
                   W + (size_t)(n0 + row1) * KT + k0 + seg1 * 8);
  };

  v8f acc[4][2];
#pragma unroll
  for (int mi = 0; mi < 4; ++mi)
#pragma unroll
    for (int ni = 0; ni < 2; ++ni) acc[mi][ni] = zero8();

  const int NT = KT / 32;
  issue_tile(0, 0);
  for (int t = 0; t < NT; ++t) {
    const int buf = t & 1;
    if (t + 1 < NT) {
      issue_tile((t + 1) * 32, (t + 1) & 1);
      wait_async_le4();  // in-order ASYNCcnt: <=4 outstanding => tile t landed
    } else {
      wait_async_0();
    }
    __syncthreads();  // all waves' tile-t async copies complete

    v16bf af[4], bfr[2];
#pragma unroll
    for (int mi = 0; mi < 4; ++mi) {
      int mr = wm * 64 + mi * 16 + l16;
      v8bf lo = *(const v8bf*)(&As[buf][mr * 32 + khalf * 8]);
      v8bf hi = *(const v8bf*)(&As[buf][mr * 32 + 16 + khalf * 8]);
      af[mi] = make_v16(lo, hi);
    }
#pragma unroll
    for (int ni = 0; ni < 2; ++ni) {
      int nr = wn * 32 + ni * 16 + l16;
      bfr[ni] = make_v16(*(const v8bf*)(&Bs[buf][nr * 32 + khalf * 8]),
                         *(const v8bf*)(&Bs[buf][nr * 32 + 16 + khalf * 8]));
    }
#pragma unroll
    for (int mi = 0; mi < 4; ++mi)
#pragma unroll
      for (int ni = 0; ni < 2; ++ni)
        acc[mi][ni] = __builtin_amdgcn_wmma_f32_16x16x32_bf16(
            false, af[mi], false, bfr[ni], (short)0, acc[mi][ni], false, false);
    __syncthreads();  // done reading buf before it is overwritten at t+2
  }

#pragma unroll
  for (int mi = 0; mi < 4; ++mi)
#pragma unroll
    for (int ni = 0; ni < 2; ++ni) {
      int nn = n0 + wn * 32 + ni * 16 + l16;
      float bv = bias[nn];
#pragma unroll
      for (int i = 0; i < 8; ++i) {
        int mm = m0 + wm * 64 + mi * 16 + khalf * 8 + i;
        float v = acc[mi][ni][i] + bv;
        if (GELU) v = 0.5f * v * (1.0f + erff(v * 0.70710678118654752f));
        size_t idx = (size_t)mm * Nout + nn;
        if (OUTF == 0) ((bf16*)outp)[idx] = (bf16)v;
        else           ((float*)outp)[idx] = v + res[idx];
      }
    }
}

// ---------------- window attention (both branches), WMMA QK^T and PV
// grid 4096 blocks of 64 threads (2 waves = 2 heads per block)
__global__ __launch_bounds__(64) void attn_kernel(
    const bf16* __restrict__ qkv, const float* __restrict__ rpb1,
    const float* __restrict__ rpb2, bf16* __restrict__ attn_out) {
  __shared__ __align__(32) bf16 vts[2][32 * 64];  // v transposed [ch][tok]
  __shared__ __align__(32) bf16 ps[2][64 * 64];   // softmax probs [m][n]

  const int hp = blockIdx.x & 3;
  const int rest = blockIdx.x >> 2;
  const int branch = rest >> 9;
  const int widx = rest & 511;
  const int b = widx >> 6, w = widx & 63;
  const int wy = w >> 3, wx = w & 7;
  const int wid = threadIdx.x >> 5, lane = threadIdx.x & 31;
  const int head = hp * 2 + wid;
  const int khalf = lane >> 4, l16 = lane & 15;
  const int cb = branch * 256 + head * 32;

  auto nglob = [&](int t) -> int {
    int gy = wy * 8 + (t >> 3), gx = wx * 8 + (t & 7);
    if (branch) { gy = (gy + 4) & 63; gx = (gx + 4) & 63; }
    return gy * 64 + gx;
  };

  // stage V transposed into LDS
#pragma unroll
  for (int t2 = 0; t2 < 2; ++t2) {
    int t = lane + t2 * 32;
    const bf16* row = qkv + (size_t)(b * NTOK + nglob(t)) * 1536 + 1024 + cb;
    v16bf v0 = *(const v16bf*)(row);
    v16bf v1 = *(const v16bf*)(row + 16);
#pragma unroll
    for (int c = 0; c < 16; ++c) {
      vts[wid][c * 64 + t] = v0[c];
      vts[wid][(16 + c) * 64 + t] = v1[c];
    }
  }

  // Q (A-frags) and K (B-frags) straight from global
  v16bf qf[4], kf[4];
#pragma unroll
  for (int mi = 0; mi < 4; ++mi) {
    int t = mi * 16 + l16;
    const bf16* row = qkv + (size_t)(b * NTOK + nglob(t)) * 1536 + cb;
    v8bf lo = *(const v8bf*)(row + khalf * 8);
    v8bf hi = *(const v8bf*)(row + 16 + khalf * 8);
    qf[mi] = make_v16(lo, hi);
  }
#pragma unroll
  for (int nj = 0; nj < 4; ++nj) {
    int t = nj * 16 + l16;
    const bf16* row = qkv + (size_t)(b * NTOK + nglob(t)) * 1536 + 512 + cb;
    kf[nj] = make_v16(*(const v8bf*)(row + khalf * 8), *(const v8bf*)(row + 16 + khalf * 8));
  }

  // S = Q K^T  (64x64, K=32)
  v8f s[4][4];
#pragma unroll
  for (int mi = 0; mi < 4; ++mi)
#pragma unroll
    for (int nj = 0; nj < 4; ++nj) s[mi][nj] = zero8();
#pragma unroll
  for (int mi = 0; mi < 4; ++mi)
#pragma unroll
    for (int nj = 0; nj < 4; ++nj)
      s[mi][nj] = __builtin_amdgcn_wmma_f32_16x16x32_bf16(
          false, qf[mi], false, kf[nj], (short)0, s[mi][nj], false, false);

  // bias + mask + softmax, write P (bf16) to LDS
  const float scale = 0.17677669529663687f;  // 1/sqrt(32)
  const float* rpb = branch ? rpb2 : rpb1;
#pragma unroll
  for (int mi = 0; mi < 4; ++mi) {
#pragma unroll
    for (int i = 0; i < 8; ++i) {
      int m = mi * 16 + khalf * 8 + i;
      int my = m >> 3, mx = m & 7;
      int gym = wy * 8 + my, gxm = wx * 8 + mx;
      int lm = 3 * (gym < 56 ? 0 : (gym < 60 ? 1 : 2)) +
                   (gxm < 56 ? 0 : (gxm < 60 ? 1 : 2));
      float vals[4];
      float rowmax = -1e30f;
#pragma unroll
      for (int nj = 0; nj < 4; ++nj) {
        int n = nj * 16 + l16;
        int ny = n >> 3, nx = n & 7;
        int ridx = (my - ny) + (mx - nx) + 29;
        float v = s[mi][nj][i] * scale + rpb[ridx * 8 + head];
        if (branch) {
          int gyn = wy * 8 + ny, gxn = wx * 8 + nx;
          int ln = 3 * (gyn < 56 ? 0 : (gyn < 60 ? 1 : 2)) +
                       (gxn < 56 ? 0 : (gxn < 60 ? 1 : 2));
          v += (lm != ln) ? -100.0f : 0.0f;
        }
        vals[nj] = v;
        rowmax = fmaxf(rowmax, v);
      }
#pragma unroll
      for (int d = 1; d < 16; d <<= 1)
        rowmax = fmaxf(rowmax, __shfl_xor(rowmax, d, 32));
      float rsum = 0.f;
#pragma unroll
      for (int nj = 0; nj < 4; ++nj) {
        vals[nj] = __expf(vals[nj] - rowmax);
        rsum += vals[nj];
      }
#pragma unroll
      for (int d = 1; d < 16; d <<= 1) rsum += __shfl_xor(rsum, d, 32);
      float inv = 1.0f / rsum;
#pragma unroll
      for (int nj = 0; nj < 4; ++nj)
        ps[wid][m * 64 + nj * 16 + l16] = (bf16)(vals[nj] * inv);
    }
  }
  __syncthreads();

  // O = P V  (64x32, K=64 -> two k-steps)
  v8f o[4][2];
#pragma unroll
  for (int mi = 0; mi < 4; ++mi)
#pragma unroll
    for (int nj = 0; nj < 2; ++nj) o[mi][nj] = zero8();
#pragma unroll
  for (int ks = 0; ks < 2; ++ks) {
    v16bf pf[4], vf[2];
#pragma unroll
    for (int mi = 0; mi < 4; ++mi) {
      int m = mi * 16 + l16;
      v8bf lo = *(const v8bf*)(&ps[wid][m * 64 + ks * 32 + khalf * 8]);
      v8bf hi = *(const v8bf*)(&ps[wid][m * 64 + ks * 32 + 16 + khalf * 8]);
      pf[mi] = make_v16(lo, hi);
    }
#pragma unroll
    for (int nj = 0; nj < 2; ++nj) {
      int n = nj * 16 + l16;
      vf[nj] = make_v16(*(const v8bf*)(&vts[wid][n * 64 + ks * 32 + khalf * 8]),
                        *(const v8bf*)(&vts[wid][n * 64 + ks * 32 + 16 + khalf * 8]));
    }
#pragma unroll
    for (int mi = 0; mi < 4; ++mi)
#pragma unroll
      for (int nj = 0; nj < 2; ++nj)
        o[mi][nj] = __builtin_amdgcn_wmma_f32_16x16x32_bf16(
            false, pf[mi], false, vf[nj], (short)0, o[mi][nj], false, false);
  }

  // scatter to attn buffer (concat x1|x2 along channels; shift undone by nglob)
#pragma unroll
  for (int mi = 0; mi < 4; ++mi)
#pragma unroll
    for (int nj = 0; nj < 2; ++nj) {
      int c = cb + nj * 16 + l16;
#pragma unroll
      for (int i = 0; i < 8; ++i) {
        int m = mi * 16 + khalf * 8 + i;
        attn_out[(size_t)(b * NTOK + nglob(m)) * DIM_ + c] = (bf16)(o[mi][nj][i]);
      }
    }
}

// ---------------- host launch
extern "C" void kernel_launch(void* const* d_in, const int* in_sizes, int n_in,
                              void* d_out, int out_size, void* d_ws, size_t ws_size,
                              hipStream_t stream) {
  const float* x      = (const float*)d_in[0];
  const float* style  = (const float*)d_in[1];
  const float* ada1_w = (const float*)d_in[2];
  const float* ada1_b = (const float*)d_in[3];
  const float* qkv_w  = (const float*)d_in[4];
  const float* qkv_b  = (const float*)d_in[5];
  const float* rpb1   = (const float*)d_in[6];
  const float* rpb2   = (const float*)d_in[7];
  const float* proj_w = (const float*)d_in[8];
  const float* proj_b = (const float*)d_in[9];
  const float* ada2_w = (const float*)d_in[10];
  const float* ada2_b = (const float*)d_in[11];
  const float* mlp_w1 = (const float*)d_in[12];
  const float* mlp_b1 = (const float*)d_in[13];
  const float* mlp_w2 = (const float*)d_in[14];
  const float* mlp_b2 = (const float*)d_in[15];

  char* ws = (char*)d_ws;
  size_t off = 0;
  auto carve = [&](size_t bytes) -> char* {
    char* p = ws + off;
    off += (bytes + 255) & ~(size_t)255;
    return p;
  };
  float* s1    = (float*)carve(8 * 1024 * 4);
  float* s2    = (float*)carve(8 * 1024 * 4);
  float* meanb = (float*)carve(8 * 512 * 4);
  float* rstdb = (float*)carve(8 * 512 * 4);
  bf16* xn     = (bf16*)carve((size_t)MTOT * 512 * 2);
  bf16* wqkv   = (bf16*)carve((size_t)1536 * 512 * 2);
  bf16* wproj  = (bf16*)carve((size_t)512 * 512 * 2);
  bf16* wm1    = (bf16*)carve((size_t)2048 * 512 * 2);
  bf16* wm2    = (bf16*)carve((size_t)512 * 2048 * 2);
  bf16* big    = (bf16*)carve((size_t)MTOT * 2048 * 2);  // qkv then h1 overlay
  bf16* attn   = (bf16*)carve((size_t)MTOT * 512 * 2);
  float* y1    = (float*)carve((size_t)MTOT * 512 * 4);
  bf16* qkvbuf = big;
  bf16* h1     = big;

  // 1. style modulations
  style_kernel<<<8, 256, 0, stream>>>(style, ada1_w, ada1_b, s1);
  style_kernel<<<8, 256, 0, stream>>>(style, ada2_w, ada2_b, s2);

  // 2. weight conversion to bf16
  cvt_kernel<<<(1536 * 512 + 255) / 256, 256, 0, stream>>>(qkv_w, wqkv, 1536 * 512);
  cvt_kernel<<<(512 * 512 + 255) / 256, 256, 0, stream>>>(proj_w, wproj, 512 * 512);
  cvt_kernel<<<(2048 * 512 + 255) / 256, 256, 0, stream>>>(mlp_w1, wm1, 2048 * 512);
  cvt_kernel<<<(512 * 2048 + 255) / 256, 256, 0, stream>>>(mlp_w2, wm2, 512 * 2048);

  // 3. adain1
  stats_kernel<<<16, 256, 0, stream>>>(x, meanb, rstdb);
  norm_kernel<<<(MTOT * 512) / 256, 256, 0, stream>>>(x, meanb, rstdb, s1, xn);

  // 4. qkv GEMM (bf16 out)
  gemm_kernel<512, false, 0><<<dim3(1536 / 128, MTOT / 128), 256, 0, stream>>>(
      xn, wqkv, qkv_b, nullptr, qkvbuf, 1536);

  // 5. window attention (both branches)
  attn_kernel<<<4096, 64, 0, stream>>>(qkvbuf, rpb1, rpb2, attn);

  // 6. proj GEMM + residual -> y1 (f32)
  gemm_kernel<512, false, 1><<<dim3(512 / 128, MTOT / 128), 256, 0, stream>>>(
      attn, wproj, proj_b, x, y1, 512);

  // 7. adain2
  stats_kernel<<<16, 256, 0, stream>>>(y1, meanb, rstdb);
  norm_kernel<<<(MTOT * 512) / 256, 256, 0, stream>>>(y1, meanb, rstdb, s2, xn);

  // 8. mlp1 GEMM + exact gelu (bf16 out, overlays dead qkv buffer)
  gemm_kernel<512, true, 0><<<dim3(2048 / 128, MTOT / 128), 256, 0, stream>>>(
      xn, wm1, mlp_b1, nullptr, h1, 2048);

  // 9. mlp2 GEMM + residual -> d_out (f32)
  gemm_kernel<2048, false, 1><<<dim3(512 / 128, MTOT / 128), 256, 0, stream>>>(
      h1, wm2, mlp_b2, y1, d_out, 512);
}
